// HiPAMA_80504866996810
// MI455X (gfx1250) — matrix operand order repaired and checked by reading, not compiled, not run on hardware
//
#include <hip/hip_runtime.h>
#include <hip/hip_bf16.h>

typedef __attribute__((ext_vector_type(16))) _Float16 v16h;
typedef __attribute__((ext_vector_type(8)))  _Float16 v8h;
typedef __attribute__((ext_vector_type(8)))  float    v8f;

#define B_   32
#define S_   512
#define E_   512
#define H_   4
#define HD_  128
#define IN_  84
#define E4_  2048
#define LSTM_NB 8

// ---------------------------------------------------------------------------
// WMMA helpers (f16 in, f32 acc).  A: 16x32 (M x K), BT: rows are N, cols K.
// ---------------------------------------------------------------------------
__device__ __forceinline__ v16h ldA16(const _Float16* p) {
    v8h lo = *(const v8h*)p;
    v8h hi = *(const v8h*)(p + 16);
    return __builtin_shufflevector(lo, hi, 0,1,2,3,4,5,6,7,8,9,10,11,12,13,14,15);
}
__device__ __forceinline__ v16h ldB16(const _Float16* p) {
    v8h lo = *(const v8h*)p;
    v8h hi = *(const v8h*)(p + 16);
    return __builtin_shufflevector(lo, hi, 0,1,2,3,4,5,6,7,8,9,10,11,12,13,14,15);
}
__device__ __forceinline__ v8f wmma16(v16h a, v16h b, v8f c) {
    return __builtin_amdgcn_wmma_f32_16x16x32_f16(false, a, false, b, (short)0, c, false, false);
}
__device__ __forceinline__ float sigm(float x) { return 1.0f / (1.0f + __expf(-x)); }

// ---------------------------------------------------------------------------
// Generic batched GEMM:  C[z] = alpha * A[z] * BT[z]^T + bias (+ C[z])
// A: [M,K] row-major f16 (lda), BT: [N,K] row-major f16 (ldb).
// Two-level batch: z -> (z/inner)*off_o + (z%inner)*off_i.
// Optional f32 C (ldc) and/or f16 mirror Ch (same indexing).
// Block: 256 thr = 8 waves, wave tile 32x32, block tile 64(M) x 128(N).
// ---------------------------------------------------------------------------
__global__ void gemm_wmma(const _Float16* __restrict__ A, long Ao, long Ai, int lda,
                          const _Float16* __restrict__ BT, long Bo, long Bi, int ldb,
                          float* __restrict__ C, _Float16* __restrict__ Ch,
                          long Co, long Ci, int ldc,
                          const float* __restrict__ bias,
                          int M, int N, int K, float alpha, int accum, int inner)
{
    int z = blockIdx.z;
    A  += (long)(z / inner) * Ao + (long)(z % inner) * Ai;
    BT += (long)(z / inner) * Bo + (long)(z % inner) * Bi;
    long co = (long)(z / inner) * Co + (long)(z % inner) * Ci;

    const int lane = threadIdx.x & 31;
    const int wave = threadIdx.x >> 5;
    const int mbase = blockIdx.y * 64 + (wave >> 2) * 32;
    const int nbase = blockIdx.x * 128 + (wave & 3) * 32;
    if (mbase >= M || nbase >= N) return;

    const int ml  = lane & 15;
    const int k0a = (lane >> 4) << 3;   // A frag: +0 / +8, second half at +16
    const int k0b = (lane >> 4) << 3;   // B frag: +0 / +8, second half at +16

    const _Float16* a0p = A + (size_t)(mbase + ml) * lda + k0a;
    const _Float16* a1p = a0p + (size_t)16 * lda;
    const _Float16* b0p = BT + (size_t)(nbase + ml) * ldb + k0b;
    const _Float16* b1p = b0p + (size_t)16 * ldb;

    v8f c00 = {}, c01 = {}, c10 = {}, c11 = {};
    for (int kb = 0; kb < K; kb += 32) {
        v16h a0 = ldA16(a0p + kb);
        v16h a1 = ldA16(a1p + kb);
        v16h b0 = ldB16(b0p + kb);
        v16h b1 = ldB16(b1p + kb);
        c00 = wmma16(a0, b0, c00);
        c01 = wmma16(a0, b1, c01);
        c10 = wmma16(a1, b0, c10);
        c11 = wmma16(a1, b1, c11);
    }

    const int rb = (lane >> 4) << 3;
    #pragma unroll
    for (int q = 0; q < 4; ++q) {
        v8f acc = (q == 0) ? c00 : (q == 1) ? c01 : (q == 2) ? c10 : c11;
        int r0 = mbase + ((q >> 1) ? 16 : 0) + rb;
        int cc = nbase + ((q & 1) ? 16 : 0) + ml;
        float bv = bias ? bias[cc] : 0.0f;
        #pragma unroll
        for (int j = 0; j < 8; ++j) {
            long ci = co + (long)(r0 + j) * ldc + cc;
            float v = acc[j] * alpha + bv;
            if (accum) v += C[ci];
            if (C)  C[ci]  = v;
            if (Ch) Ch[ci] = (_Float16)v;
        }
    }
}

// ---------------------------------------------------------------------------
// Persistent LSTM layer: 8 blocks x 8 waves = 64 waves, each one 32x32 tile
// of the [B=32, 4E=2048] gate matrix.  Grid barrier between WMMA phase and
// elementwise cell update, looping over t.
// ---------------------------------------------------------------------------
__device__ __forceinline__ void gsync(int* bar, int* epoch) {
    __syncthreads();
    if (threadIdx.x == 0) {
        ++(*epoch);
        __threadfence();
        atomicAdd(bar, 1);
        int target = (*epoch) * LSTM_NB;
        volatile int* vb = (volatile int*)bar;
        while (*vb < target) { __builtin_amdgcn_s_sleep(1); }
    }
    __syncthreads();
    __threadfence();
}

__global__ void lstm_layer(const float* __restrict__ gx,      // [B*S, 4E]
                           const _Float16* __restrict__ whhT, // [4E, E] f16
                           _Float16* __restrict__ y16,        // [B,S,E] f16
                           float* cstate, _Float16* hstate,   // [B,E]
                           float* gates,                      // [B,4E]
                           int* bar)
{
    const int lane = threadIdx.x & 31;
    const int wave = threadIdx.x >> 5;
    const int gwave = blockIdx.x * 8 + wave;        // 0..63
    const int nbase = gwave * 32;
    const int gtid  = blockIdx.x * 256 + threadIdx.x;
    const int tot   = LSTM_NB * 256;
    const int ml  = lane & 15;
    const int k0a = (lane >> 4) << 3;
    const int k0b = (lane >> 4) << 3;
    const int rb  = (lane >> 4) << 3;
    int epoch = 0;

    for (int t = 0; t < S_; ++t) {
        v8f c00 = {}, c01 = {}, c10 = {}, c11 = {};
        const _Float16* a0p = hstate + (size_t)ml * E_ + k0a;
        const _Float16* a1p = a0p + (size_t)16 * E_;
        const _Float16* b0p = whhT + (size_t)(nbase + ml) * E_ + k0b;
        const _Float16* b1p = b0p + (size_t)16 * E_;
        for (int kb = 0; kb < E_; kb += 32) {
            v16h a0 = ldA16(a0p + kb);
            v16h a1 = ldA16(a1p + kb);
            v16h b0 = ldB16(b0p + kb);
            v16h b1 = ldB16(b1p + kb);
            c00 = wmma16(a0, b0, c00);
            c01 = wmma16(a0, b1, c01);
            c10 = wmma16(a1, b0, c10);
            c11 = wmma16(a1, b1, c11);
        }
        #pragma unroll
        for (int j = 0; j < 8; ++j) {
            int r = rb + j;                // batch row for c00/c01; +16 for c10/c11
            long gr0 = ((long)r * S_ + t) * E4_;
            long gr1 = ((long)(r + 16) * S_ + t) * E4_;
            gates[(size_t)r * E4_ + nbase + ml]            = c00[j] + gx[gr0 + nbase + ml];
            gates[(size_t)r * E4_ + nbase + 16 + ml]       = c01[j] + gx[gr0 + nbase + 16 + ml];
            gates[(size_t)(r + 16) * E4_ + nbase + ml]     = c10[j] + gx[gr1 + nbase + ml];
            gates[(size_t)(r + 16) * E4_ + nbase + 16 + ml]= c11[j] + gx[gr1 + nbase + 16 + ml];
        }
        gsync(bar, &epoch);
        for (int i = gtid; i < B_ * E_; i += tot) {
            int b = i >> 9, e = i & 511;
            float gi = gates[(size_t)b * E4_ + e];
            float gf = gates[(size_t)b * E4_ + E_ + e];
            float gg = gates[(size_t)b * E4_ + 2 * E_ + e];
            float go = gates[(size_t)b * E4_ + 3 * E_ + e];
            float c  = sigm(gf) * cstate[i] + sigm(gi) * tanhf(gg);
            float h  = sigm(go) * tanhf(c);
            cstate[i] = c;
            hstate[i] = (_Float16)h;
            y16[((size_t)b * S_ + t) * E_ + e] = (_Float16)h;
        }
        gsync(bar, &epoch);
    }
}

// ---------------------------------------------------------------------------
// Small utility kernels
// ---------------------------------------------------------------------------
__global__ void embed_kernel(const float* __restrict__ x, const int* __restrict__ phn,
                             const float* __restrict__ w_phn, const float* __restrict__ b_phn,
                             const float* __restrict__ w_in, const float* __restrict__ b_in,
                             _Float16* __restrict__ h16)
{
    int row = blockIdx.x;                       // b*S + s
    __shared__ float xr[IN_];
    if (threadIdx.x < IN_) xr[threadIdx.x] = x[(size_t)row * IN_ + threadIdx.x];
    __syncthreads();
    int p = phn[row] + 1;
    for (int e = threadIdx.x; e < E_; e += 256) {
        float acc = b_in[e] + b_phn[e] + w_phn[(size_t)p * E_ + e];
        for (int k = 0; k < IN_; ++k) acc += xr[k] * w_in[(size_t)k * E_ + e];
        h16[(size_t)row * E_ + e] = (_Float16)acc;
    }
}

__global__ void softmax_rows(const float* __restrict__ X, _Float16* __restrict__ P, int W)
{
    long r = blockIdx.x;
    const float* x = X + r * (long)W;
    _Float16* p = P + r * (long)W;
    __shared__ float red[256];
    float mx = -1e30f;
    for (int i = threadIdx.x; i < W; i += 256) mx = fmaxf(mx, x[i]);
    red[threadIdx.x] = mx; __syncthreads();
    for (int s = 128; s > 0; s >>= 1) { if (threadIdx.x < s) red[threadIdx.x] = fmaxf(red[threadIdx.x], red[threadIdx.x + s]); __syncthreads(); }
    mx = red[0]; __syncthreads();
    float sum = 0.0f;
    for (int i = threadIdx.x; i < W; i += 256) { float e = __expf(x[i] - mx); p[i] = (_Float16)e; sum += e; }
    red[threadIdx.x] = sum; __syncthreads();
    for (int s = 128; s > 0; s >>= 1) { if (threadIdx.x < s) red[threadIdx.x] += red[threadIdx.x + s]; __syncthreads(); }
    float inv = 1.0f / red[0];
    __syncthreads();
    for (int i = threadIdx.x; i < W; i += 256) p[i] = (_Float16)((float)p[i] * inv);
}

__global__ void head_kernel(const float* __restrict__ X, const float* __restrict__ g,
                            const float* __restrict__ b, const float* __restrict__ w,
                            const float* __restrict__ bb, float* __restrict__ out, int nrows)
{
    int row = blockIdx.x * 8 + (threadIdx.x >> 5);
    int lane = threadIdx.x & 31;
    if (row >= nrows) return;
    const float* x = X + (size_t)row * E_;
    float s = 0.0f, s2 = 0.0f;
    float xs[16];
    #pragma unroll
    for (int i = 0; i < 16; ++i) { float v = x[i * 32 + lane]; xs[i] = v; s += v; s2 += v * v; }
    for (int m = 16; m > 0; m >>= 1) { s += __shfl_xor(s, m, 32); s2 += __shfl_xor(s2, m, 32); }
    float mean = s * (1.0f / E_);
    float var  = s2 * (1.0f / E_) - mean * mean;
    float rr   = rsqrtf(var + 1e-5f);
    float d = 0.0f;
    #pragma unroll
    for (int i = 0; i < 16; ++i) { int e = i * 32 + lane; d += ((xs[i] - mean) * rr * g[e] + b[e]) * w[e]; }
    for (int m = 16; m > 0; m >>= 1) d += __shfl_xor(d, m, 32);
    if (lane == 0) out[row] = d + bb[0];
}

__global__ void tcvt_kernel(const float* __restrict__ in, _Float16* __restrict__ out,
                            int R, int C, long si, long so)
{
    long z = blockIdx.y;
    const float* ip = in + z * si;
    _Float16* op = out + z * so;
    int idx = blockIdx.x * 256 + threadIdx.x;
    if (idx >= R * C) return;
    int r = idx / C, c = idx % C;
    op[(long)c * R + r] = (_Float16)ip[idx];
}

__global__ void trans_h(const _Float16* __restrict__ in, _Float16* __restrict__ out,
                        int R, int C, int ldi, int ldo,
                        long si_o, long si_i, long so_o, long so_i, int inner)
{
    int z = blockIdx.y;
    const _Float16* ip = in + (long)(z / inner) * si_o + (long)(z % inner) * si_i;
    _Float16* op = out + (long)(z / inner) * so_o + (long)(z % inner) * so_i;
    int idx = blockIdx.x * 256 + threadIdx.x;
    if (idx >= R * C) return;
    int r = idx / C, c = idx % C;
    op[(long)c * ldo + r] = ip[(long)r * ldi + c];
}

__global__ void cvt32to16(const float* __restrict__ in, _Float16* __restrict__ out, long n)
{ long i = (long)blockIdx.x * 256 + threadIdx.x; if (i < n) out[i] = (_Float16)in[i]; }

__global__ void cvt16to32(const _Float16* __restrict__ in, float* __restrict__ out, long n)
{ long i = (long)blockIdx.x * 256 + threadIdx.x; if (i < n) out[i] = (float)in[i]; }

__global__ void add2_kernel(const float* a, const float* b, float* o, int n)
{ int i = blockIdx.x * 256 + threadIdx.x; if (i < n) o[i] = a[i] + b[i]; }

__global__ void convw_prep(const float* __restrict__ w, _Float16* __restrict__ out)
{
    int idx = blockIdx.x * 256 + threadIdx.x;
    if (idx >= 5 * E_ * E_) return;
    int i = idx & 511, o = (idx >> 9) & 511, dk = idx >> 18;
    out[idx] = (_Float16)w[((size_t)(o * E_ + i)) * 5 + dk];
}

__global__ void pad_copy(const _Float16* __restrict__ in, _Float16* __restrict__ out)
{
    long idx = (long)blockIdx.x * 256 + threadIdx.x;
    if (idx >= (long)B_ * (S_ + 4) * E_) return;
    int e = idx & 511;
    long t = idx >> 9;
    int s = (int)(t % (S_ + 4));
    int b = (int)(t / (S_ + 4));
    _Float16 v = (_Float16)0.0f;
    if (s >= 2 && s < S_ + 2) v = in[((long)b * S_ + (s - 2)) * E_ + e];
    out[idx] = v;
}

__global__ void rep_avg(const float* a, const float* b, const float* c, _Float16* out, long n)
{
    long i = (long)blockIdx.x * 256 + threadIdx.x;
    if (i < n) out[i] = (_Float16)((a[i] + b[i] + c[i]) * (1.0f / 3.0f));
}

__global__ void last_valid(const int* __restrict__ phn, const float* __restrict__ hsU,
                           float* __restrict__ out)
{
    int b = threadIdx.x;
    if (b >= B_) return;
    int idx = S_ - 1;
    for (int s = 0; s < S_; ++s) { if (phn[b * S_ + s] < 0) { idx = s - 1; break; } }
    if (idx < 0) idx = S_ - 1;
    for (int i = 0; i < 5; ++i) out[i * B_ + b] = hsU[(long)i * B_ * S_ + (long)b * S_ + idx];
}

// ---------------------------------------------------------------------------
// Host orchestration
// ---------------------------------------------------------------------------
typedef _Float16 h16;

static void launch_gemm(hipStream_t st, const h16* A, long Ao, long Ai, int lda,
                        const h16* BT, long Bo, long Bi, int ldb,
                        float* C, h16* Ch, long Co, long Ci, int ldc,
                        const float* bias, int M, int N, int K, float alpha,
                        int accum, int inner, int nbatch)
{
    dim3 grid((N + 127) / 128, (M + 63) / 64, nbatch), block(256);
    gemm_wmma<<<grid, block, 0, st>>>(A, Ao, Ai, lda, BT, Bo, Bi, ldb,
                                      C, Ch, Co, Ci, ldc, bias, M, N, K, alpha, accum, inner);
}

extern "C" void kernel_launch(void* const* d_in, const int* in_sizes, int n_in,
                              void* d_out, int out_size, void* d_ws, size_t ws_size,
                              hipStream_t stream)
{
    const float* x        = (const float*)d_in[0];
    const int*   phn      = (const int*)d_in[1];
    const float* w_phn    = (const float*)d_in[2];
    const float* b_phn    = (const float*)d_in[3];
    const float* w_in     = (const float*)d_in[4];
    const float* b_in     = (const float*)d_in[5];
    const float* lstm_wih = (const float*)d_in[6];
    const float* lstm_whh = (const float*)d_in[7];
    const float* lstm_bih = (const float*)d_in[8];
    const float* lstm_bhh = (const float*)d_in[9];
    const float* attn_qkv = (const float*)d_in[10];
    const float* attn_pw  = (const float*)d_in[11];
    const float* attn_pb  = (const float*)d_in[12];
    const float* conv_w   = (const float*)d_in[13];
    const float* conv_b   = (const float*)d_in[14];
    const float* wattn_qkv= (const float*)d_in[15];
    const float* wattn_pw = (const float*)d_in[16];
    const float* wattn_pb = (const float*)d_in[17];
    const float* rep_w_w  = (const float*)d_in[18];
    const float* rep_w_b  = (const float*)d_in[19];
    const float* rep_u_w  = (const float*)d_in[20];
    const float* rep_u_b  = (const float*)d_in[21];
    const float* ln_g     = (const float*)d_in[22];
    const float* ln_b     = (const float*)d_in[23];
    const float* head_w   = (const float*)d_in[24];
    const float* head_b   = (const float*)d_in[25];
    float* outp = (float*)d_out;

    const float SCALE  = 0.08838834764831845f;  // HD^-0.5
    const float ASCALE = 0.04419417382415922f;  // E^-0.5
    const long  BSE = (long)B_ * S_ * E_;       // 8,388,608

    // ---- bump allocator over d_ws --------------------------------------
    size_t off = 0;
    auto alloc = [&](size_t bytes) -> void* {
        off = (off + 255) & ~(size_t)255;
        void* p = (char*)d_ws + off;
        off += bytes;
        return p;
    };
    h16*  wih16   = (h16*)alloc((size_t)2 * E4_ * E_ * 2);
    h16*  whh16   = (h16*)alloc((size_t)2 * E4_ * E_ * 2);
    float* combb  = (float*)alloc((size_t)2 * E4_ * 4);
    h16*  qkvT    = (h16*)alloc((size_t)3 * E_ * E_ * 2);
    h16*  projT   = (h16*)alloc((size_t)E_ * E_ * 2);
    h16*  wqkvT   = (h16*)alloc((size_t)3 * E_ * E_ * 2);
    h16*  wprojT  = (h16*)alloc((size_t)E_ * E_ * 2);
    h16*  convT   = (h16*)alloc((size_t)5 * E_ * E_ * 2);
    h16*  repwT   = (h16*)alloc((size_t)3 * E_ * E_ * 2);
    h16*  repuT   = (h16*)alloc((size_t)5 * E_ * E_ * 2);
    h16*  e16A    = (h16*)alloc((size_t)BSE * 2);
    h16*  e16B    = (h16*)alloc((size_t)BSE * 2);
    float* gxbuf  = (float*)alloc((size_t)B_ * S_ * E4_ * 4);  // also attention scores
    h16*  P16     = (h16*)alloc((size_t)B_ * S_ * E4_ * 2);
    h16*  qkv16   = (h16*)alloc((size_t)B_ * S_ * 3 * E_ * 2);
    h16*  Vt16    = (h16*)alloc((size_t)B_ * H_ * HD_ * S_ * 2);
    h16*  xpad    = (h16*)alloc((size_t)B_ * (S_ + 4) * E_ * 2);
    float* convo  = (float*)alloc((size_t)BSE * 4);
    h16*  ws16    = (h16*)alloc((size_t)3 * BSE * 2);
    h16*  wsT16   = (h16*)alloc((size_t)3 * BSE * 2);
    h16*  us16    = (h16*)alloc((size_t)5 * BSE * 2);
    h16*  usT16   = (h16*)alloc((size_t)5 * BSE * 2);
    float* wattn  = (float*)alloc((size_t)3 * BSE * 4);
    float* uattn  = (float*)alloc((size_t)BSE * 4);
    float* hsU    = (float*)alloc((size_t)5 * B_ * S_ * 4);
    float* cstate = (float*)alloc((size_t)B_ * E_ * 4);
    h16*  hstate  = (h16*)alloc((size_t)B_ * E_ * 2);
    float* gates  = (float*)alloc((size_t)B_ * E4_ * 4);
    int*  bar     = (int*)alloc(256);
    (void)ws_size; (void)n_in; (void)in_sizes; (void)out_size;

    float* scores = gxbuf;  // reuse gate buffer for attention scores

    // ---- weight prep ----------------------------------------------------
    {
        long n = (long)2 * E4_ * E_;
        cvt32to16<<<(n + 255) / 256, 256, 0, stream>>>(lstm_wih, wih16, n);
        cvt32to16<<<(n + 255) / 256, 256, 0, stream>>>(lstm_whh, whh16, n);
        add2_kernel<<<(2 * E4_ + 255) / 256, 256, 0, stream>>>(lstm_bih, lstm_bhh, combb, 2 * E4_);
        dim3 g1(((E_ * 3 * E_) + 255) / 256, 1);
        tcvt_kernel<<<g1, 256, 0, stream>>>(attn_qkv,  qkvT,  E_, 3 * E_, 0, 0);
        tcvt_kernel<<<g1, 256, 0, stream>>>(wattn_qkv, wqkvT, E_, 3 * E_, 0, 0);
        dim3 g2(((E_ * E_) + 255) / 256, 1);
        tcvt_kernel<<<g2, 256, 0, stream>>>(attn_pw,  projT,  E_, E_, 0, 0);
        tcvt_kernel<<<g2, 256, 0, stream>>>(wattn_pw, wprojT, E_, E_, 0, 0);
        dim3 g3(((E_ * E_) + 255) / 256, 3);
        tcvt_kernel<<<g3, 256, 0, stream>>>(rep_w_w, repwT, E_, E_, (long)E_ * E_, (long)E_ * E_);
        dim3 g5(((E_ * E_) + 255) / 256, 5);
        tcvt_kernel<<<g5, 256, 0, stream>>>(rep_u_w, repuT, E_, E_, (long)E_ * E_, (long)E_ * E_);
        convw_prep<<<((5 * E_ * E_) + 255) / 256, 256, 0, stream>>>(conv_w, convT);
    }

    // ---- embed ----------------------------------------------------------
    embed_kernel<<<B_ * S_, 256, 0, stream>>>(x, phn, w_phn, b_phn, w_in, b_in, e16A);

    // ---- LSTM layer 0 ---------------------------------------------------
    launch_gemm(stream, e16A, 0, 0, E_, wih16, 0, 0, E_,
                gxbuf, nullptr, 0, 0, E4_, combb, B_ * S_, E4_, E_, 1.0f, 0, 1, 1);
    hipMemsetAsync(cstate, 0, (size_t)B_ * E_ * 4, stream);
    hipMemsetAsync(hstate, 0, (size_t)B_ * E_ * 2, stream);
    hipMemsetAsync(bar, 0, 256, stream);
    lstm_layer<<<LSTM_NB, 256, 0, stream>>>(gxbuf, whh16, e16B, cstate, hstate, gates, bar);
    // ---- LSTM layer 1 ---------------------------------------------------
    launch_gemm(stream, e16B, 0, 0, E_, wih16 + (size_t)E4_ * E_, 0, 0, E_,
                gxbuf, nullptr, 0, 0, E4_, combb + E4_, B_ * S_, E4_, E_, 1.0f, 0, 1, 1);
    hipMemsetAsync(cstate, 0, (size_t)B_ * E_ * 4, stream);
    hipMemsetAsync(hstate, 0, (size_t)B_ * E_ * 2, stream);
    hipMemsetAsync(bar, 0, 256, stream);
    lstm_layer<<<LSTM_NB, 256, 0, stream>>>(gxbuf, whh16 + (size_t)E4_ * E_, e16A, cstate, hstate, gates, bar);

    // ---- MHA helper (shared by both attentions) -------------------------
    auto run_mha = [&](const h16* in16, const h16* qkvW, const h16* projW,
                       const float* projB, h16* tmp16, h16* out16) {
        // qkv: [B*S, 1536] f16
        launch_gemm(stream, in16, 0, 0, E_, qkvW, 0, 0, E_,
                    nullptr, qkv16, 0, 0, 3 * E_, nullptr, B_ * S_, 3 * E_, E_, 1.0f, 0, 1, 1);
        // V^T per (b,h): [HD, S]
        {
            dim3 g(((S_ * HD_) + 255) / 256, B_ * H_);
            trans_h<<<g, 256, 0, stream>>>(qkv16 + 2 * E_, Vt16, S_, HD_, 3 * E_, S_,
                                           (long)S_ * 3 * E_, HD_, (long)H_ * HD_ * S_, (long)HD_ * S_, H_);
        }
        // scores[b,h] = Q K^T * SCALE
        launch_gemm(stream, qkv16, (long)S_ * 3 * E_, HD_, 3 * E_,
                    qkv16 + E_, (long)S_ * 3 * E_, HD_, 3 * E_,
                    scores, nullptr, (long)H_ * S_ * S_, (long)S_ * S_, S_,
                    nullptr, S_, S_, HD_, SCALE, 0, H_, B_ * H_);
        softmax_rows<<<B_ * H_ * S_, 256, 0, stream>>>(scores, P16, S_);
        // out[b,s,h*HD+:] = P V
        launch_gemm(stream, P16, (long)H_ * S_ * S_, (long)S_ * S_, S_,
                    Vt16, (long)H_ * HD_ * S_, (long)HD_ * S_, S_,
                    nullptr, tmp16, (long)S_ * E_, HD_, E_,
                    nullptr, S_, HD_, S_, 1.0f, 0, H_, B_ * H_);
        // proj
        launch_gemm(stream, tmp16, 0, 0, E_, projW, 0, 0, E_,
                    nullptr, out16, 0, 0, E_, projB, B_ * S_, E_, E_, 1.0f, 0, 1, 1);
    };

    // ---- first MHA: e16A -> e16A (via e16B temp) ------------------------
    run_mha(e16A, qkvT, projT, attn_pb, e16B, e16A);

    // ---- conv1d as 5 accumulated GEMMs ----------------------------------
    pad_copy<<<(((long)B_ * (S_ + 4) * E_) + 255) / 256, 256, 0, stream>>>(e16A, xpad);
    for (int dk = 0; dk < 5; ++dk) {
        launch_gemm(stream, xpad + (size_t)dk * E_, (long)(S_ + 4) * E_, 0, E_,
                    convT + (size_t)dk * E_ * E_, 0, 0, E_,
                    convo, (dk == 4) ? e16B : nullptr, (long)S_ * E_, 0, E_,
                    (dk == 0) ? conv_b : nullptr, S_, E_, E_, 1.0f, (dk > 0) ? 1 : 0, 1, B_);
    }
    // p head (index 0) -> d_out[160 .. 160+16384)
    head_kernel<<<(B_ * S_ + 7) / 8, 256, 0, stream>>>(convo, ln_g, ln_b, head_w, head_b,
                                                       outp + 5 * B_, B_ * S_);

    // ---- rep_w projections + transposes ---------------------------------
    for (int i = 0; i < 3; ++i) {
        launch_gemm(stream, e16B, 0, 0, E_, repwT + (size_t)i * E_ * E_, 0, 0, E_,
                    nullptr, ws16 + (size_t)i * BSE, 0, 0, E_,
                    rep_w_b + i * E_, B_ * S_, E_, E_, 1.0f, 0, 1, 1);
        dim3 g(((S_ * E_) + 255) / 256, B_);
        trans_h<<<g, 256, 0, stream>>>(ws16 + (size_t)i * BSE, wsT16 + (size_t)i * BSE,
                                       S_, E_, E_, S_, (long)S_ * E_, 0, (long)E_ * S_, 0, 1);
    }
    // ---- w aspect attentions --------------------------------------------
    for (int i = 0; i < 3; ++i) {
        int js[2], nj = 0;
        for (int j = 0; j < 3; ++j) if (j != i) js[nj++] = j;
        for (int jj = 0; jj < 2; ++jj)
            launch_gemm(stream, ws16 + (size_t)i * BSE, (long)S_ * E_, 0, E_,
                        ws16 + (size_t)js[jj] * BSE, (long)S_ * E_, 0, E_,
                        scores + (size_t)jj * S_, nullptr, (long)S_ * 2 * S_, 0, 2 * S_,
                        nullptr, S_, S_, E_, ASCALE, 0, 1, B_);
        softmax_rows<<<B_ * S_, 256, 0, stream>>>(scores, P16, 2 * S_);
        cvt16to32<<<(BSE + 255) / 256, 256, 0, stream>>>(ws16 + (size_t)i * BSE,
                                                         wattn + (size_t)i * BSE, BSE);
        for (int jj = 0; jj < 2; ++jj)
            launch_gemm(stream, P16 + (size_t)jj * S_, (long)S_ * 2 * S_, 0, 2 * S_,
                        wsT16 + (size_t)js[jj] * BSE, (long)E_ * S_, 0, S_,
                        wattn + (size_t)i * BSE, nullptr, (long)S_ * E_, 0, E_,
                        nullptr, S_, E_, S_, 1.0f, 1, 1, B_);
        // w head (index 1+i) -> d_out[160+(1+i)*16384 ...)
        head_kernel<<<(B_ * S_ + 7) / 8, 256, 0, stream>>>(wattn + (size_t)i * BSE,
            ln_g + (1 + i) * E_, ln_b + (1 + i) * E_, head_w + (1 + i) * E_, head_b + (1 + i),
            outp + 5 * B_ + (size_t)(1 + i) * B_ * S_, B_ * S_);
    }

    // ---- rep = mean(w_attn) then second MHA -----------------------------
    rep_avg<<<(BSE + 255) / 256, 256, 0, stream>>>(wattn, wattn + BSE, wattn + 2 * BSE, e16A, BSE);
    run_mha(e16A, wqkvT, wprojT, wattn_pb, e16B, e16A);

    // ---- rep_u projections + transposes ---------------------------------
    for (int i = 0; i < 5; ++i) {
        launch_gemm(stream, e16A, 0, 0, E_, repuT + (size_t)i * E_ * E_, 0, 0, E_,
                    nullptr, us16 + (size_t)i * BSE, 0, 0, E_,
                    rep_u_b + i * E_, B_ * S_, E_, E_, 1.0f, 0, 1, 1);
        dim3 g(((S_ * E_) + 255) / 256, B_);
        trans_h<<<g, 256, 0, stream>>>(us16 + (size_t)i * BSE, usT16 + (size_t)i * BSE,
                                       S_, E_, E_, S_, (long)S_ * E_, 0, (long)E_ * S_, 0, 1);
    }
    // ---- u aspect attentions + heads ------------------------------------
    for (int i = 0; i < 5; ++i) {
        int js[4], nj = 0;
        for (int j = 0; j < 5; ++j) if (j != i) js[nj++] = j;
        for (int jj = 0; jj < 4; ++jj)
            launch_gemm(stream, us16 + (size_t)i * BSE, (long)S_ * E_, 0, E_,
                        us16 + (size_t)js[jj] * BSE, (long)S_ * E_, 0, E_,
                        scores + (size_t)jj * S_, nullptr, (long)S_ * 4 * S_, 0, 4 * S_,
                        nullptr, S_, S_, E_, ASCALE, 0, 1, B_);
        softmax_rows<<<B_ * S_, 256, 0, stream>>>(scores, P16, 4 * S_);
        cvt16to32<<<(BSE + 255) / 256, 256, 0, stream>>>(us16 + (size_t)i * BSE, uattn, BSE);
        for (int jj = 0; jj < 4; ++jj)
            launch_gemm(stream, P16 + (size_t)jj * S_, (long)S_ * 4 * S_, 0, 4 * S_,
                        usT16 + (size_t)js[jj] * BSE, (long)E_ * S_, 0, S_,
                        uattn, nullptr, (long)S_ * E_, 0, E_,
                        nullptr, S_, E_, S_, 1.0f, 1, 1, B_);
        // u head (index 4+i) -> hsU scratch
        head_kernel<<<(B_ * S_ + 7) / 8, 256, 0, stream>>>(uattn,
            ln_g + (4 + i) * E_, ln_b + (4 + i) * E_, head_w + (4 + i) * E_, head_b + (4 + i),
            hsU + (size_t)i * B_ * S_, B_ * S_);
    }

    // ---- last-valid gather -> d_out[0..160) -----------------------------
    last_valid<<<1, 32, 0, stream>>>(phn, hsU, outp);
}
